// GPT2QuantBlock_33801392619893
// MI455X (gfx1250) — compile-verified, matrix operand order checked by reading, not run on hardware
//
#include <hip/hip_runtime.h>
#include <stdint.h>

constexpr int Bb   = 2;
constexpr int Sq   = 2048;
constexpr int Dd   = 1024;
constexpr int Hn   = 16;
constexpr int HD   = 64;
constexpr int IN   = 4096;
constexpr int Mtok = Bb * Sq;
constexpr float EPS = 1e-5f;

typedef int      v8i_t  __attribute__((ext_vector_type(8)));
typedef float    v8f_t  __attribute__((ext_vector_type(8)));
typedef _Float16 v16h_t __attribute__((ext_vector_type(16)));

union V16H { v16h_t v; uint4 q[2]; };
union V8I  { v8i_t  v; unsigned long long q[4]; };

__global__ __launch_bounds__(256) void absmax_k(const float* __restrict__ x,
                                                size_t n, unsigned* __restrict__ amax) {
  __shared__ float red[256];
  float lm = 0.f;
  for (size_t i = (size_t)blockIdx.x * 256 + threadIdx.x; i < n;
       i += (size_t)gridDim.x * 256)
    lm = fmaxf(lm, fabsf(x[i]));
  red[threadIdx.x] = lm;
  __syncthreads();
  for (int off = 128; off > 0; off >>= 1) {
    if ((int)threadIdx.x < off) red[threadIdx.x] = fmaxf(red[threadIdx.x], red[threadIdx.x + off]);
    __syncthreads();
  }
  if (threadIdx.x == 0) atomicMax(amax, __float_as_uint(red[0]));
}

__global__ __launch_bounds__(256) void quantize_k(const float* __restrict__ x,
                                                  size_t n,
                                                  const unsigned* __restrict__ amax,
                                                  int8_t* __restrict__ q) {
  const float s   = fmaxf(__uint_as_float(*amax), 1e-8f) / 127.0f;
  const float inv = 1.0f / s;
  for (size_t i = (size_t)blockIdx.x * 256 + threadIdx.x; i < n;
       i += (size_t)gridDim.x * 256) {
    float t = x[i] * inv;
    t = fminf(fmaxf(t, -127.0f), 127.0f);
    q[i] = (int8_t)(int)rintf(t);
  }
}

__global__ __launch_bounds__(256) void layernorm_absmax_k(
    const float* __restrict__ x, const float* __restrict__ g,
    const float* __restrict__ be, float* __restrict__ y,
    unsigned* __restrict__ amax, int Dc) {
  __shared__ float red[256];
  const int tid = threadIdx.x;
  const float* xr = x + (size_t)blockIdx.x * Dc;
  float s = 0.f;
  for (int i = tid; i < Dc; i += 256) s += xr[i];
  red[tid] = s; __syncthreads();
  for (int off = 128; off > 0; off >>= 1) {
    if (tid < off) red[tid] += red[tid + off];
    __syncthreads();
  }
  const float mean = red[0] / (float)Dc;
  __syncthreads();
  float v = 0.f;
  for (int i = tid; i < Dc; i += 256) { float d = xr[i] - mean; v += d * d; }
  red[tid] = v; __syncthreads();
  for (int off = 128; off > 0; off >>= 1) {
    if (tid < off) red[tid] += red[tid + off];
    __syncthreads();
  }
  const float rstd = rsqrtf(red[0] / (float)Dc + EPS);
  __syncthreads();
  float* yr = y + (size_t)blockIdx.x * Dc;
  float lm = 0.f;
  for (int i = tid; i < Dc; i += 256) {
    float o = (xr[i] - mean) * rstd * g[i] + be[i];
    yr[i] = o;
    lm = fmaxf(lm, fabsf(o));
  }
  red[tid] = lm; __syncthreads();
  for (int off = 128; off > 0; off >>= 1) {
    if (tid < off) red[tid] = fmaxf(red[tid], red[tid + off]);
    __syncthreads();
  }
  if (tid == 0) atomicMax(amax, __float_as_uint(red[0]));
}

template <int OMODE, bool GELU, bool RES, bool AMAX>
__global__ __launch_bounds__(256) void gemm_i8_k(
    const int8_t* __restrict__ Aq, const int8_t* __restrict__ Wq,
    const unsigned* __restrict__ amaxA, const unsigned* __restrict__ amaxW,
    const float* __restrict__ bias, const float* __restrict__ res,
    void* __restrict__ outv, unsigned* __restrict__ outAmax, int K, int N) {
  __shared__ int8_t bsm[64][80];
  const int tid  = threadIdx.x;
  const int lane = tid & 31;
  const int wave = tid >> 5;
  const int half = lane >> 4;
  const int l15  = lane & 15;
  const int m0   = blockIdx.y * 128 + wave * 16;
  const int n0   = blockIdx.x * 64;

  const int8_t* arow = Aq + (size_t)(m0 + l15) * K;
  const int brow = tid >> 2;
  const int bcol = (tid & 3) << 4;
  const int8_t* wsrc = Wq + (size_t)(n0 + brow) * K + bcol;

  *(uint4*)&bsm[brow][bcol] = *(const uint4*)wsrc;
  __syncthreads();

  v8i_t acc[4] = {};
  for (int kb = 0; kb < K; kb += 64) {
    const bool more = (kb + 64) < K;
    uint4 nxt;
    if (more) {
      nxt = *(const uint4*)(wsrc + kb + 64);
      __builtin_prefetch(arow + kb + 64, 0, 0);
    }
    V8I a;
#pragma unroll
    for (int c = 0; c < 4; ++c)
      a.q[c] = *(const unsigned long long*)(arow + kb + c * 16 + (half << 3));
#pragma unroll
    for (int t = 0; t < 4; ++t) {
      const int8_t* bl = &bsm[t * 16 + l15][0];
      V8I bfrag;
#pragma unroll
      for (int c = 0; c < 4; ++c)
        bfrag.q[c] = *(const unsigned long long*)(bl + c * 16 + (half << 3));
      acc[t] = __builtin_amdgcn_wmma_i32_16x16x64_iu8(true, a.v, true, bfrag.v,
                                                      acc[t], false, false);
    }
    if (more) {
      __syncthreads();
      *(uint4*)&bsm[brow][bcol] = nxt;
      __syncthreads();
    }
  }

  const float sA = fmaxf(__uint_as_float(*amaxA), 1e-8f) / 127.0f;
  const float sW = fmaxf(__uint_as_float(*amaxW), 1e-8f) / 127.0f;
  const float alpha = sA * sW;
  float lm = 0.f;
#pragma unroll
  for (int t = 0; t < 4; ++t) {
    const int col = n0 + t * 16 + l15;
    const float bv = bias[col];
#pragma unroll
    for (int r = 0; r < 8; ++r) {
      const int orow = m0 + r + 8 * half;
      float v = alpha * (float)acc[t][r] + bv;
      if (OMODE == 0) {
        if (GELU) v = 0.5f * v * (1.0f + erff(v * 0.70710678118654752f));
        if (RES) v += res[(size_t)orow * N + col];
        ((float*)outv)[(size_t)orow * N + col] = v;
        if (AMAX) lm = fmaxf(lm, fabsf(v));
      } else if (OMODE == 1) {
        ((_Float16*)outv)[(size_t)orow * N + col] = (_Float16)v;
      } else {
        const int bi = orow >> 11, sidx = orow & (Sq - 1);
        const int hh = col >> 6, hd = col & 63;
        ((_Float16*)outv)[(((size_t)(bi * Hn + hh) * 64 + hd) << 11) + sidx] =
            (_Float16)v;
      }
    }
  }
  if (AMAX) {
#pragma unroll
    for (int m = 16; m >= 1; m >>= 1) lm = fmaxf(lm, __shfl_xor(lm, m));
    if (lane == 0) atomicMax(outAmax, __float_as_uint(lm));
  }
}

__global__ __launch_bounds__(128) void attention_k(
    const _Float16* __restrict__ Q, const _Float16* __restrict__ Km,
    const _Float16* __restrict__ Vt, const float* __restrict__ mask,
    float* __restrict__ O, unsigned* __restrict__ amax) {
  __shared__ _Float16 pbuf[4][32][48];
  const int lane = threadIdx.x & 31;
  const int wave = threadIdx.x >> 5;
  const int half = lane >> 4;
  const int h8   = half << 3;
  const int l15  = lane & 15;
  const int S32  = Sq / 32;
  const int tile = blockIdx.x * 4 + wave;
  const int b  = tile / (Hn * S32);
  const int h  = (tile / S32) % Hn;
  const int qb = tile % S32;

  V16H aQ[2][2];
#pragma unroll
  for (int qt = 0; qt < 2; ++qt) {
    const _Float16* qrow =
        Q + ((size_t)b * Sq + qb * 32 + qt * 16 + l15) * Dd + h * HD;
    aQ[qt][0].q[0] = *(const uint4*)(qrow + h8);
    aQ[qt][0].q[1] = *(const uint4*)(qrow + 16 + h8);
    aQ[qt][1].q[0] = *(const uint4*)(qrow + 32 + h8);
    aQ[qt][1].q[1] = *(const uint4*)(qrow + 48 + h8);
  }

  float mrun[2][8], lrun[2][8];
#pragma unroll
  for (int qt = 0; qt < 2; ++qt)
#pragma unroll
    for (int r = 0; r < 8; ++r) { mrun[qt][r] = -1e30f; lrun[qt][r] = 0.f; }
  v8f_t o[2][4] = {};

  for (int kb = 0; kb < Sq; kb += 32) {
    V16H bK[2][2];
    float madd[2];
#pragma unroll
    for (int st = 0; st < 2; ++st) {
      const int key = kb + st * 16 + l15;
      const _Float16* krow = Km + ((size_t)b * Sq + key) * Dd + h * HD;
      bK[st][0].q[0] = *(const uint4*)(krow + h8);
      bK[st][0].q[1] = *(const uint4*)(krow + 16 + h8);
      bK[st][1].q[0] = *(const uint4*)(krow + 32 + h8);
      bK[st][1].q[1] = *(const uint4*)(krow + 48 + h8);
      madd[st] = (1.0f - mask[(size_t)b * Sq + key]) * -10000.0f;
    }
#pragma unroll
    for (int qt = 0; qt < 2; ++qt) {
      v8f_t s[2];
#pragma unroll
      for (int st = 0; st < 2; ++st) {
        v8f_t c = {};
        c = __builtin_amdgcn_wmma_f32_16x16x32_f16(false, aQ[qt][0].v, false,
                                                   bK[st][0].v, (short)0, c,
                                                   false, false);
        c = __builtin_amdgcn_wmma_f32_16x16x32_f16(false, aQ[qt][1].v, false,
                                                   bK[st][1].v, (short)0, c,
                                                   false, false);
#pragma unroll
        for (int r = 0; r < 8; ++r) c[r] = c[r] * 0.125f + madd[st];
        s[st] = c;
      }
      float fac[8];
#pragma unroll
      for (int r = 0; r < 8; ++r) {
        float rm = fmaxf(s[0][r], s[1][r]);
#pragma unroll
        for (int m = 8; m >= 1; m >>= 1) rm = fmaxf(rm, __shfl_xor(rm, m));
        const float mn = fmaxf(mrun[qt][r], rm);
        fac[r] = expf(mrun[qt][r] - mn);
        const float p0 = expf(s[0][r] - mn);
        const float p1 = expf(s[1][r] - mn);
        s[0][r] = p0; s[1][r] = p1;
        float rs = p0 + p1;
#pragma unroll
        for (int m = 8; m >= 1; m >>= 1) rs += __shfl_xor(rs, m);
        lrun[qt][r] = lrun[qt][r] * fac[r] + rs;
        mrun[qt][r] = mn;
      }
#pragma unroll
      for (int t = 0; t < 4; ++t)
#pragma unroll
        for (int r = 0; r < 8; ++r) o[qt][t][r] *= fac[r];
#pragma unroll
      for (int r = 0; r < 8; ++r) {
        const int rr = qt * 16 + r + 8 * half;
        pbuf[wave][rr][l15]      = (_Float16)s[0][r];
        pbuf[wave][rr][16 + l15] = (_Float16)s[1][r];
      }
    }
    asm volatile("s_wait_dscnt 0" ::: "memory");

    V16H bV[4];
#pragma unroll
    for (int t = 0; t < 4; ++t) {
      const _Float16* vrow =
          Vt + (((size_t)(b * Hn + h) * 64 + t * 16 + l15) << 11) + kb;
      bV[t].q[0] = *(const uint4*)(vrow + h8);
      bV[t].q[1] = *(const uint4*)(vrow + 16 + h8);
    }
#pragma unroll
    for (int qt = 0; qt < 2; ++qt) {
      V16H pA;
      const _Float16* prow = &pbuf[wave][qt * 16 + l15][0];
      pA.q[0] = *(const uint4*)(prow + h8);
      pA.q[1] = *(const uint4*)(prow + 16 + h8);
#pragma unroll
      for (int t = 0; t < 4; ++t)
        o[qt][t] = __builtin_amdgcn_wmma_f32_16x16x32_f16(
            false, pA.v, false, bV[t].v, (short)0, o[qt][t], false, false);
    }
  }

  float lm = 0.f;
#pragma unroll
  for (int qt = 0; qt < 2; ++qt)
#pragma unroll
    for (int t = 0; t < 4; ++t)
#pragma unroll
      for (int r = 0; r < 8; ++r) {
        const float v = o[qt][t][r] / lrun[qt][r];
        O[((size_t)b * Sq + qb * 32 + qt * 16 + r + 8 * half) * Dd + h * HD +
          t * 16 + l15] = v;
        lm = fmaxf(lm, fabsf(v));
      }
#pragma unroll
  for (int m = 16; m >= 1; m >>= 1) lm = fmaxf(lm, __shfl_xor(lm, m));
  if (lane == 0) atomicMax(amax, __float_as_uint(lm));
}

extern "C" void kernel_launch(void* const* d_in, const int* in_sizes, int n_in,
                              void* d_out, int out_size, void* d_ws, size_t ws_size,
                              hipStream_t stream) {
  (void)in_sizes; (void)n_in; (void)out_size; (void)ws_size;
  const float* hidden = (const float*)d_in[0];
  const float* mask   = (const float*)d_in[1];
  const float* wq = (const float*)d_in[2];  const float* bq = (const float*)d_in[3];
  const float* wk = (const float*)d_in[4];  const float* bk = (const float*)d_in[5];
  const float* wv = (const float*)d_in[6];  const float* bv = (const float*)d_in[7];
  const float* wo = (const float*)d_in[8];  const float* bo = (const float*)d_in[9];
  const float* w1 = (const float*)d_in[10]; const float* b1 = (const float*)d_in[11];
  const float* w2 = (const float*)d_in[12]; const float* b2 = (const float*)d_in[13];
  const float* g1 = (const float*)d_in[14]; const float* be1 = (const float*)d_in[15];
  const float* g2 = (const float*)d_in[16]; const float* be2 = (const float*)d_in[17];
  float* out = (float*)d_out;

  char* ws = (char*)d_ws;
  unsigned* sc = (unsigned*)ws;
  enum { SC_LN1 = 0, SC_WQ, SC_WK, SC_WV, SC_WO, SC_W1, SC_W2, SC_ATT, SC_LN2, SC_GEL };
  size_t off = 256;
  float*     lnbuf = (float*)(ws + off);     off += (size_t)Mtok * Dd * 4;
  int8_t*    xq    = (int8_t*)(ws + off);    off += (size_t)Mtok * Dd;
  int8_t*    wqq   = (int8_t*)(ws + off);    off += (size_t)Dd * Dd;
  int8_t*    wkq   = (int8_t*)(ws + off);    off += (size_t)Dd * Dd;
  int8_t*    wvq   = (int8_t*)(ws + off);    off += (size_t)Dd * Dd;
  int8_t*    woq   = (int8_t*)(ws + off);    off += (size_t)Dd * Dd;
  int8_t*    w1q   = (int8_t*)(ws + off);    off += (size_t)IN * Dd;
  int8_t*    w2q   = (int8_t*)(ws + off);    off += (size_t)Dd * IN;
  _Float16*  qh    = (_Float16*)(ws + off);  off += (size_t)Mtok * Dd * 2;
  _Float16*  kh    = (_Float16*)(ws + off);  off += (size_t)Mtok * Dd * 2;
  _Float16*  vT    = (_Float16*)(ws + off);  off += (size_t)Mtok * Dd * 2;
  float*     h2    = (float*)(ws + off);     off += (size_t)Mtok * Dd * 4;
  float*     gel   = (float*)(ws + off);     off += (size_t)Mtok * IN * 4;
  int8_t*    hq    = (int8_t*)(ws + off);    off += (size_t)Mtok * IN;

  hipMemsetAsync(sc, 0, 256, stream);

  absmax_k<<<256, 256, 0, stream>>>(wq, (size_t)Dd * Dd, sc + SC_WQ);
  absmax_k<<<256, 256, 0, stream>>>(wk, (size_t)Dd * Dd, sc + SC_WK);
  absmax_k<<<256, 256, 0, stream>>>(wv, (size_t)Dd * Dd, sc + SC_WV);
  absmax_k<<<256, 256, 0, stream>>>(wo, (size_t)Dd * Dd, sc + SC_WO);
  absmax_k<<<512, 256, 0, stream>>>(w1, (size_t)IN * Dd, sc + SC_W1);
  absmax_k<<<512, 256, 0, stream>>>(w2, (size_t)Dd * IN, sc + SC_W2);
  quantize_k<<<1024, 256, 0, stream>>>(wq, (size_t)Dd * Dd, sc + SC_WQ, wqq);
  quantize_k<<<1024, 256, 0, stream>>>(wk, (size_t)Dd * Dd, sc + SC_WK, wkq);
  quantize_k<<<1024, 256, 0, stream>>>(wv, (size_t)Dd * Dd, sc + SC_WV, wvq);
  quantize_k<<<1024, 256, 0, stream>>>(wo, (size_t)Dd * Dd, sc + SC_WO, woq);
  quantize_k<<<2048, 256, 0, stream>>>(w1, (size_t)IN * Dd, sc + SC_W1, w1q);
  quantize_k<<<2048, 256, 0, stream>>>(w2, (size_t)Dd * IN, sc + SC_W2, w2q);

  layernorm_absmax_k<<<Mtok, 256, 0, stream>>>(hidden, g1, be1, lnbuf, sc + SC_LN1, Dd);
  quantize_k<<<1024, 256, 0, stream>>>(lnbuf, (size_t)Mtok * Dd, sc + SC_LN1, xq);
  dim3 gD(Dd / 64, Mtok / 128);
  gemm_i8_k<1, false, false, false><<<gD, 256, 0, stream>>>(
      xq, wqq, sc + SC_LN1, sc + SC_WQ, bq, nullptr, (void*)qh, nullptr, Dd, Dd);
  gemm_i8_k<1, false, false, false><<<gD, 256, 0, stream>>>(
      xq, wkq, sc + SC_LN1, sc + SC_WK, bk, nullptr, (void*)kh, nullptr, Dd, Dd);
  gemm_i8_k<2, false, false, false><<<gD, 256, 0, stream>>>(
      xq, wvq, sc + SC_LN1, sc + SC_WV, bv, nullptr, (void*)vT, nullptr, Dd, Dd);

  attention_k<<<(Bb * Hn * (Sq / 32)) / 4, 128, 0, stream>>>(qh, kh, vT, mask,
                                                             lnbuf, sc + SC_ATT);
  quantize_k<<<1024, 256, 0, stream>>>(lnbuf, (size_t)Mtok * Dd, sc + SC_ATT, xq);
  gemm_i8_k<0, false, true, false><<<gD, 256, 0, stream>>>(
      xq, woq, sc + SC_ATT, sc + SC_WO, bo, hidden, (void*)h2, nullptr, Dd, Dd);

  layernorm_absmax_k<<<Mtok, 256, 0, stream>>>(h2, g2, be2, lnbuf, sc + SC_LN2, Dd);
  quantize_k<<<1024, 256, 0, stream>>>(lnbuf, (size_t)Mtok * Dd, sc + SC_LN2, xq);
  dim3 g1d(IN / 64, Mtok / 128);
  gemm_i8_k<0, true, false, true><<<g1d, 256, 0, stream>>>(
      xq, w1q, sc + SC_LN2, sc + SC_W1, b1, nullptr, (void*)gel, sc + SC_GEL, Dd, IN);
  quantize_k<<<4096, 256, 0, stream>>>(gel, (size_t)Mtok * IN, sc + SC_GEL, hq);
  gemm_i8_k<0, false, true, false><<<gD, 256, 0, stream>>>(
      hq, w2q, sc + SC_GEL, sc + SC_W2, b2, h2, (void*)out, nullptr, IN, Dd);
}
